// MultiModalClinicalGraphSAGE_67757404062358
// MI455X (gfx1250) — matrix operand order repaired and checked by reading, not compiled, not run on hardware
//
#include <hip/hip_runtime.h>
#include <stdint.h>

// ---------------- problem constants (from reference) ----------------
#define N_NODES 100000
#define CLIN_DIM 64
#define MEL_DIM 1280
#define H_DIM 128
#define CAT_DIM (H_DIM + CLIN_DIM)   // 192
#define NCLASS 4
#define NTILES_N (H_DIM / 16)        // 8 column tiles per 16-row block

typedef __attribute__((ext_vector_type(16))) __bf16 v16bf;
typedef __attribute__((ext_vector_type(8)))  __bf16 v8bf;
typedef __attribute__((ext_vector_type(8)))  float  v8f;

// ---------------- helpers ----------------
__device__ __forceinline__ __bf16 f2bf_scalar(float f) {   // weight cvt only
  union { float f; unsigned u; } x; x.f = f;
  unsigned r = x.u + 0x7FFFu + ((x.u >> 16) & 1u);
  unsigned short h = (unsigned short)(r >> 16);
  __bf16 b;
  __builtin_memcpy(&b, &h, sizeof(b));
  return b;
}

__device__ __forceinline__ v16bf cat8(v8bf lo, v8bf hi) {
  return __builtin_shufflevector(lo, hi, 0, 1, 2, 3, 4, 5, 6, 7,
                                 8, 9, 10, 11, 12, 13, 14, 15);
}

// A fragment: 16x32 bf16 tile, contiguous k layout (B uses the same):
//   lane 0-15 -> M=lane (K base +0), lane 16-31 -> M=lane-16 (K base +16)
//   half h -> k = kb + h  => two adjacent 8-float runs.
// Two 32-byte v8f loads + packed v_cvt to bf16.
__device__ __forceinline__ v16bf load_a_frag(const float* __restrict__ rowp, int k0) {
  int lane = threadIdx.x & 31;
  int kb = k0 + ((lane >> 4) << 4);
  v8f lo = *(const v8f*)(rowp + kb);
  v8f hi = *(const v8f*)(rowp + kb + 8);
  return cat8(__builtin_convertvector(lo, v8bf),
              __builtin_convertvector(hi, v8bf));
}

__device__ __forceinline__ v16bf load_a_frag_scaled(const float* __restrict__ rowp,
                                                    int k0, float s) {
  int lane = threadIdx.x & 31;
  int kb = k0 + ((lane >> 4) << 4);
  v8f lo = *(const v8f*)(rowp + kb) * s;
  v8f hi = *(const v8f*)(rowp + kb + 8) * s;
  return cat8(__builtin_convertvector(lo, v8bf),
              __builtin_convertvector(hi, v8bf));
}

// B fragment: 32x16 tile of W^T (W row-major [out][in] bf16).
//   col n = n0 + (lane&15), k = k0 + (lane>=16 ? 16 : 0) + h
// => one contiguous 32-byte v16bf load per lane.
__device__ __forceinline__ v16bf load_b_frag(const __bf16* __restrict__ W,
                                             int n0, int k0, int ldw) {
  int lane = threadIdx.x & 31;
  return *(const v16bf*)(W + (size_t)(n0 + (lane & 15)) * ldw
                           + (k0 + ((lane >> 4) << 4)));
}

__device__ __forceinline__ v8f wmma_bf16(v16bf a, v16bf b, v8f c) {
  return __builtin_amdgcn_wmma_f32_16x16x32_bf16(false, a, false, b,
                                                 (short)0, c, false, false);
}

// D store: VGPR r -> M = row0 + r + (lane>=16 ? 8:0), N = col0 + (lane&15).
__device__ __forceinline__ void store_d(float* __restrict__ out, int ldo,
                                        int row0, int col0,
                                        const float* __restrict__ bias,
                                        v8f acc, bool relu) {
  int lane = threadIdx.x & 31;
  int n = col0 + (lane & 15);
  float bv = bias[n];
  int mb = row0 + ((lane >> 4) << 3);
#pragma unroll
  for (int r = 0; r < 8; ++r) {
    float v = acc[r] + bv;
    if (relu) v = fmaxf(v, 0.f);
    out[(size_t)(mb + r) * ldo + n] = v;
  }
}

// ---------------- kernels ----------------

__global__ void cvt_f32_to_bf16(const float* __restrict__ in,
                                __bf16* __restrict__ out, int n) {
  int i = blockIdx.x * blockDim.x + threadIdx.x;
  int stride = gridDim.x * blockDim.x;
  for (; i < n; i += stride) out[i] = f2bf_scalar(in[i]);
}

__global__ void zero_f32(float* __restrict__ p, long long n) {
  long long i = (long long)blockIdx.x * blockDim.x + threadIdx.x;
  long long stride = (long long)gridDim.x * blockDim.x;
  for (; i < n; i += stride) p[i] = 0.f;
}

// mel_h = relu(mel @ Wm^T + bm) : [N,1280] x [1280,128]
// One wave owns a 16-row block and ALL 8 column tiles: each A fragment is
// loaded once and reused by 8 WMMAs (mel is the dominant HBM stream).
__global__ void mel_gemm_kernel(const float* __restrict__ mel,
                                const __bf16* __restrict__ Wm,
                                const float* __restrict__ bm,
                                float* __restrict__ out) {
  int wid = (int)((blockIdx.x * blockDim.x + threadIdx.x) >> 5);
  int row0 = wid * 16;
  if (row0 >= N_NODES) return;
  int lane = threadIdx.x & 31;
  const float* rowp = mel + (size_t)(row0 + (lane & 15)) * MEL_DIM;
  v8f acc[NTILES_N];
#pragma unroll
  for (int t = 0; t < NTILES_N; ++t) acc[t] = (v8f){};
  for (int k0 = 0; k0 < MEL_DIM; k0 += 32) {
    v16bf a = load_a_frag(rowp, k0);
#pragma unroll
    for (int t = 0; t < NTILES_N; ++t) {
      v16bf b = load_b_frag(Wm, t * 16, k0, MEL_DIM);
      acc[t] = wmma_bf16(a, b, acc[t]);
    }
  }
#pragma unroll
  for (int t = 0; t < NTILES_N; ++t)
    store_d(out, H_DIM, row0, t * 16, bm, acc[t], true);
}

// x0 = relu(concat(clinical, mel_h) @ Wc^T + bc) : [N,192] x [192,128]
__global__ void concat_gemm_kernel(const float* __restrict__ clinical,
                                   const float* __restrict__ melh,
                                   const __bf16* __restrict__ Wc,
                                   const float* __restrict__ bc,
                                   float* __restrict__ out) {
  int wid = (int)((blockIdx.x * blockDim.x + threadIdx.x) >> 5);
  int row0 = wid * 16;
  if (row0 >= N_NODES) return;
  int lane = threadIdx.x & 31;
  int m = row0 + (lane & 15);
  const float* crow = clinical + (size_t)m * CLIN_DIM;
  const float* hrow = melh + (size_t)m * H_DIM;
  v8f acc[NTILES_N];
#pragma unroll
  for (int t = 0; t < NTILES_N; ++t) acc[t] = (v8f){};
#pragma unroll
  for (int k0 = 0; k0 < CLIN_DIM; k0 += 32) {         // k in [0,64): clinical
    v16bf a = load_a_frag(crow, k0);
#pragma unroll
    for (int t = 0; t < NTILES_N; ++t)
      acc[t] = wmma_bf16(a, load_b_frag(Wc, t * 16, k0, CAT_DIM), acc[t]);
  }
#pragma unroll
  for (int k0 = CLIN_DIM; k0 < CAT_DIM; k0 += 32) {   // k in [64,192): mel_h
    v16bf a = load_a_frag(hrow, k0 - CLIN_DIM);
#pragma unroll
    for (int t = 0; t < NTILES_N; ++t)
      acc[t] = wmma_bf16(a, load_b_frag(Wc, t * 16, k0, CAT_DIM), acc[t]);
  }
#pragma unroll
  for (int t = 0; t < NTILES_N; ++t)
    store_d(out, H_DIM, row0, t * 16, bc, acc[t], true);
}

// edge scatter: sum[dst] += x[src] (float4 chunks), optional degree count.
// Working set (x + sum + cnt ~ 103 MB) fits the 192 MB L2 -> atomics stay in L2.
__global__ void aggregate_kernel(const float* __restrict__ x,
                                 const int* __restrict__ src,
                                 const int* __restrict__ dst,
                                 float* __restrict__ sum,
                                 float* __restrict__ cnt,
                                 long long n_edges, int do_count) {
  long long tid = (long long)blockIdx.x * blockDim.x + threadIdx.x;
  if (tid >= n_edges * 32) return;
  int e = (int)(tid >> 5);
  int c = (int)(tid & 31) << 2;                 // 4-float chunk
  int s = src[e], d = dst[e];
  const float4 v = *(const float4*)(x + (size_t)s * H_DIM + c);
  float* o = sum + (size_t)d * H_DIM + c;
  atomicAdd(o + 0, v.x);
  atomicAdd(o + 1, v.y);
  atomicAdd(o + 2, v.z);
  atomicAdd(o + 3, v.w);
  if (do_count && c == 0) atomicAdd(cnt + d, 1.0f);
}

// h1 = relu((agg/cnt) @ Wl^T + bl + x0 @ Wr^T) : two [N,128]x[128,128] GEMMs
__global__ void sage1_gemm_kernel(const float* __restrict__ agg,
                                  const float* __restrict__ cnt,
                                  const float* __restrict__ x0,
                                  const __bf16* __restrict__ Wl,
                                  const float* __restrict__ bl,
                                  const __bf16* __restrict__ Wr,
                                  float* __restrict__ out) {
  int wid = (int)((blockIdx.x * blockDim.x + threadIdx.x) >> 5);
  int row0 = wid * 16;
  if (row0 >= N_NODES) return;
  int lane = threadIdx.x & 31;
  int m = row0 + (lane & 15);
  float inv = 1.0f / fmaxf(cnt[m], 1.0f);       // mean normalization fused into A
  const float* arow = agg + (size_t)m * H_DIM;
  const float* xrow = x0 + (size_t)m * H_DIM;
  v8f acc[NTILES_N];
#pragma unroll
  for (int t = 0; t < NTILES_N; ++t) acc[t] = (v8f){};
#pragma unroll
  for (int k0 = 0; k0 < H_DIM; k0 += 32) {
    v16bf a = load_a_frag_scaled(arow, k0, inv);
#pragma unroll
    for (int t = 0; t < NTILES_N; ++t)
      acc[t] = wmma_bf16(a, load_b_frag(Wl, t * 16, k0, H_DIM), acc[t]);
  }
#pragma unroll
  for (int k0 = 0; k0 < H_DIM; k0 += 32) {
    v16bf a = load_a_frag(xrow, k0);
#pragma unroll
    for (int t = 0; t < NTILES_N; ++t)
      acc[t] = wmma_bf16(a, load_b_frag(Wr, t * 16, k0, H_DIM), acc[t]);
  }
#pragma unroll
  for (int t = 0; t < NTILES_N; ++t)
    store_d(out, H_DIM, row0, t * 16, bl, acc[t], true);
}

// head: out[n,c] = (agg2[n]/cnt[n]) . Wl2[c] + bl2[c] + h1[n] . Wr2[c]
// Only 4 output columns -> WMMA would waste 12/16 N-lanes; VALU dot is right.
__global__ void sage2_head_kernel(const float* __restrict__ agg,
                                  const float* __restrict__ cnt,
                                  const float* __restrict__ x,
                                  const float* __restrict__ Wl2,
                                  const float* __restrict__ bl2,
                                  const float* __restrict__ Wr2,
                                  float* __restrict__ out) {
  int t = blockIdx.x * blockDim.x + threadIdx.x;
  if (t >= N_NODES * NCLASS) return;
  int n = t >> 2, c = t & 3;
  float inv = 1.0f / fmaxf(cnt[n], 1.0f);
  const float* a  = agg + (size_t)n * H_DIM;
  const float* xr = x   + (size_t)n * H_DIM;
  const float* wl = Wl2 + c * H_DIM;
  const float* wr = Wr2 + c * H_DIM;
  float s = bl2[c];
#pragma unroll 4
  for (int k = 0; k < H_DIM; ++k) s += a[k] * inv * wl[k] + xr[k] * wr[k];
  out[t] = s;
}

// ---------------- workspace layout (bytes) ----------------
#define OFF_WM   ((size_t)0)                       // 128*1280 bf16 = 327680
#define OFF_WC   ((size_t)327680)                  // 128*192  bf16 = 49152
#define OFF_WL1  ((size_t)376832)                  // 128*128  bf16 = 32768
#define OFF_WR1  ((size_t)409600)                  // 128*128  bf16 = 32768
#define OFF_B0   ((size_t)442368)                  // [N,128] f32 (mel_h, then h1)
#define OFF_B1   (OFF_B0 + (size_t)N_NODES * H_DIM * 4)   // [N,128] f32 (x0)
#define OFF_B2   (OFF_B1 + (size_t)N_NODES * H_DIM * 4)   // [N,128] f32 (agg)
#define OFF_CNT  (OFF_B2 + (size_t)N_NODES * H_DIM * 4)   // [N] f32 degree

extern "C" void kernel_launch(void* const* d_in, const int* in_sizes, int n_in,
                              void* d_out, int out_size, void* d_ws, size_t ws_size,
                              hipStream_t stream) {
  const float* clinical = (const float*)d_in[0];
  const float* mel      = (const float*)d_in[1];
  const int*   eidx     = (const int*)d_in[2];
  const float* Wm  = (const float*)d_in[3];
  const float* bm  = (const float*)d_in[4];
  const float* Wc  = (const float*)d_in[5];
  const float* bc  = (const float*)d_in[6];
  const float* Wl1 = (const float*)d_in[7];
  const float* bl1 = (const float*)d_in[8];
  const float* Wr1 = (const float*)d_in[9];
  const float* Wl2 = (const float*)d_in[10];
  const float* bl2 = (const float*)d_in[11];
  const float* Wr2 = (const float*)d_in[12];
  float* out = (float*)d_out;

  const long long E = in_sizes[2] / 2;
  const int* src = eidx;
  const int* dst = eidx + E;

  char* ws = (char*)d_ws;
  __bf16* Wm_bf  = (__bf16*)(ws + OFF_WM);
  __bf16* Wc_bf  = (__bf16*)(ws + OFF_WC);
  __bf16* Wl1_bf = (__bf16*)(ws + OFF_WL1);
  __bf16* Wr1_bf = (__bf16*)(ws + OFF_WR1);
  float* B0  = (float*)(ws + OFF_B0);
  float* B1  = (float*)(ws + OFF_B1);
  float* B2  = (float*)(ws + OFF_B2);
  float* cnt = (float*)(ws + OFF_CNT);

  // GEMM grid: one wave per 16-row block (N/16 = 6250 waves)
  const int gemm_waves  = (N_NODES + 15) / 16;
  const int gemm_blocks = (gemm_waves * 32 + 255) / 256;

  // 1) weight conversion f32 -> bf16 (small, every call: deterministic)
  cvt_f32_to_bf16<<<256, 256, 0, stream>>>(Wm,  Wm_bf,  H_DIM * MEL_DIM);
  cvt_f32_to_bf16<<<96,  256, 0, stream>>>(Wc,  Wc_bf,  H_DIM * CAT_DIM);
  cvt_f32_to_bf16<<<64,  256, 0, stream>>>(Wl1, Wl1_bf, H_DIM * H_DIM);
  cvt_f32_to_bf16<<<64,  256, 0, stream>>>(Wr1, Wr1_bf, H_DIM * H_DIM);

  // 2) zero agg-sum + degree counts (contiguous: N*128 + N floats)
  zero_f32<<<2048, 256, 0, stream>>>(B2, (long long)N_NODES * H_DIM + N_NODES);

  // 3) mel linear (WMMA bf16)
  mel_gemm_kernel<<<gemm_blocks, 256, 0, stream>>>(mel, Wm_bf, bm, B0);

  // 4) concat linear (WMMA bf16)
  concat_gemm_kernel<<<gemm_blocks, 256, 0, stream>>>(clinical, B0, Wc_bf, bc, B1);

  // 5) edge aggregation #1 (with degree counting)
  {
    long long threads = E * 32;
    int blocks = (int)((threads + 255) / 256);
    aggregate_kernel<<<blocks, 256, 0, stream>>>(B1, src, dst, B2, cnt, E, 1);
  }

  // 6) SAGE layer 1 (WMMA bf16, mean-normalize fused into A load) -> B0
  sage1_gemm_kernel<<<gemm_blocks, 256, 0, stream>>>(B2, cnt, B1, Wl1_bf, bl1,
                                                     Wr1_bf, B0);

  // 7) re-zero agg-sum (degree counts reused — same edge list)
  zero_f32<<<2048, 256, 0, stream>>>(B2, (long long)N_NODES * H_DIM);

  // 8) edge aggregation #2
  {
    long long threads = E * 32;
    int blocks = (int)((threads + 255) / 256);
    aggregate_kernel<<<blocks, 256, 0, stream>>>(B0, src, dst, B2, cnt, E, 0);
  }

  // 9) classifier head
  {
    int blocks = (N_NODES * NCLASS + 255) / 256;
    sage2_head_kernel<<<blocks, 256, 0, stream>>>(B2, cnt, B0, Wl2, bl2, Wr2, out);
  }
}
